// MultiHeadAttention_81716047773870
// MI455X (gfx1250) — matrix operand order repaired and checked by reading, not compiled, not run on hardware
//
#include <hip/hip_runtime.h>
#include <cstdint>

// Problem constants (match reference)
#define N_B 2
#define T_S 2048
#define E_D 1024
#define H_N 16
#define DH_ 64

typedef __bf16 bf16_t;
typedef __attribute__((ext_vector_type(16))) __bf16 v16bf;
typedef __attribute__((ext_vector_type(8)))  float  v8f;

static __device__ __forceinline__ v8f wmma_bf16(v16bf a, v16bf b, v8f c) {
  // D = A(16x32 bf16) * B(32x16 bf16) + C(16x16 f32)
  return __builtin_amdgcn_wmma_f32_16x16x32_bf16(false, a, false, b, (short)0, c,
                                                 false, false);
}

// A-fragment: 16x32 bf16, source row-major with leading dim `ld` (elements).
// Lane L<16: row M=L, K = 0..7 and 16..23 ; Lane L>=16: row M=L-16, K = 8..15 and 24..31.
static __device__ __forceinline__ v16bf load_frag_a(const bf16_t* base, int ld) {
  const int lane = threadIdx.x & 31;
  const int m    = lane & 15;
  const int kb   = (lane >> 4) ? 8 : 0;
  const bf16_t* p = base + (size_t)m * ld + kb;
  union { uint4 u[2]; v16bf v; } r;
  r.u[0] = *reinterpret_cast<const uint4*>(p);        // K = kb .. kb+7
  r.u[1] = *reinterpret_cast<const uint4*>(p + 16);   // K = kb+16 .. kb+23
  return r.v;
}

// B-fragment: 32x16 bf16 (KxN). Source stored as [N][K] row-major, leading dim `ld`.
// Lane L<16: col N=L, K = 0..7 and 16..23 ; Lane L>=16: col N=L-16, K = 8..15 and 24..31.
static __device__ __forceinline__ v16bf load_frag_b(const bf16_t* base, int ld) {
  const int lane = threadIdx.x & 31;
  const int n    = lane & 15;
  const int kb   = (lane >> 4) ? 8 : 0;
  const bf16_t* p = base + (size_t)n * ld + kb;
  union { uint4 u[2]; v16bf v; } r;
  r.u[0] = *reinterpret_cast<const uint4*>(p);        // K = kb .. kb+7
  r.u[1] = *reinterpret_cast<const uint4*>(p + 16);   // K = kb+16 .. kb+23
  return r.v;
}

// ---------------- conversion kernels ----------------
__global__ void cvt_f32_bf16(const float* __restrict__ src, bf16_t* __restrict__ dst,
                             int n) {
  int i = blockIdx.x * blockDim.x + threadIdx.x;
  if (i < n) dst[i] = (bf16_t)src[i];
}

// W: [H][E][DH] fp32  ->  Wt: [H][DH][E] bf16 (so B-fragments are contiguous)
__global__ void cvt_w_transpose(const float* __restrict__ W, bf16_t* __restrict__ Wt) {
  int i = blockIdx.x * blockDim.x + threadIdx.x; // over H*E*DH
  if (i >= H_N * E_D * DH_) return;
  int d = i % DH_;
  int e = (i / DH_) % E_D;
  int h = i / (DH_ * E_D);
  Wt[((size_t)h * DH_ + d) * E_D + e] = (bf16_t)W[i];
}

// ---------------- projection GEMM ----------------
// grid = (T/128, H, N), block = 128 (4 waves); each wave computes a 32x64 output
// tile (2 row-tiles x 4 col-tiles = 8 accumulators). Ping-pong double buffering:
// stage-B fragments load while stage-A WMMAs execute and vice versa.
__global__ void __launch_bounds__(128, 1)
proj_gemm_kernel(const bf16_t* __restrict__ X,
                 const bf16_t* __restrict__ Wt,
                 bf16_t* __restrict__ out, int transposeOut) {
  const int wave = threadIdx.x >> 5;
  const int lane = threadIdx.x & 31;
  const int hf = lane >> 4;
  const int nl = lane & 15;
  const int h = blockIdx.y, n = blockIdx.z;
  const int m0 = blockIdx.x * 128 + wave * 32;

  const bf16_t* Xn = X + (size_t)n * T_S * E_D;
  const bf16_t* Wh = Wt + (size_t)h * DH_ * E_D;
  const bf16_t* A0 = Xn + (size_t)m0 * E_D;
  const bf16_t* A1 = Xn + (size_t)(m0 + 16) * E_D;

  v8f acc0[4] = {{}, {}, {}, {}};
  v8f acc1[4] = {{}, {}, {}, {}};

  // stage A fragments (e = 0)
  v16bf a0A = load_frag_a(A0, E_D);
  v16bf a1A = load_frag_a(A1, E_D);
  v16bf bA[4];
#pragma unroll
  for (int nt = 0; nt < 4; ++nt) bA[nt] = load_frag_b(Wh + (size_t)(nt * 16) * E_D, E_D);

  for (int e = 0; e < E_D; e += 64) {
    const int e1 = e + 32;
    // stage B fragment loads overlap stage A WMMAs
    v16bf a0B = load_frag_a(A0 + e1, E_D);
    v16bf a1B = load_frag_a(A1 + e1, E_D);
    v16bf bB[4];
#pragma unroll
    for (int nt = 0; nt < 4; ++nt)
      bB[nt] = load_frag_b(Wh + (size_t)(nt * 16) * E_D + e1, E_D);
#pragma unroll
    for (int nt = 0; nt < 4; ++nt) {
      acc0[nt] = wmma_bf16(a0A, bA[nt], acc0[nt]);
      acc1[nt] = wmma_bf16(a1A, bA[nt], acc1[nt]);
    }
    const int e2 = e + 64;
    if (e2 < E_D) {  // stage A fragment loads overlap stage B WMMAs
      a0A = load_frag_a(A0 + e2, E_D);
      a1A = load_frag_a(A1 + e2, E_D);
#pragma unroll
      for (int nt = 0; nt < 4; ++nt)
        bA[nt] = load_frag_b(Wh + (size_t)(nt * 16) * E_D + e2, E_D);
    }
#pragma unroll
    for (int nt = 0; nt < 4; ++nt) {
      acc0[nt] = wmma_bf16(a0B, bB[nt], acc0[nt]);
      acc1[nt] = wmma_bf16(a1B, bB[nt], acc1[nt]);
    }
  }

  if (transposeOut == 0) {
    // row-major [T][DH] per (n,h)
    bf16_t* O = out + ((size_t)(n * H_N + h)) * T_S * DH_;
#pragma unroll
    for (int r = 0; r < 8; ++r) {
      const size_t r0 = (size_t)(m0 + r + hf * 8);
      const size_t r1 = r0 + 16;
#pragma unroll
      for (int nt = 0; nt < 4; ++nt) {
        O[r0 * DH_ + nt * 16 + nl] = (bf16_t)acc0[nt][r];
        O[r1 * DH_ + nt * 16 + nl] = (bf16_t)acc1[nt][r];
      }
    }
  } else {
    // transposed [DH][T] per (n,h) (for V): the 8 rows a lane owns are
    // consecutive, so pack and store as b128.
    bf16_t* O = out + ((size_t)(n * H_N + h)) * DH_ * T_S;
    const size_t rbase = (size_t)(m0 + hf * 8);
#pragma unroll
    for (int nt = 0; nt < 4; ++nt) {
      union { bf16_t hh[8]; uint4 u; } pk0, pk1;
#pragma unroll
      for (int r = 0; r < 8; ++r) {
        pk0.hh[r] = (bf16_t)acc0[nt][r];
        pk1.hh[r] = (bf16_t)acc1[nt][r];
      }
      bf16_t* col = O + (size_t)(nt * 16 + nl) * T_S;
      *reinterpret_cast<uint4*>(col + rbase)      = pk0.u;
      *reinterpret_cast<uint4*>(col + rbase + 16) = pk1.u;
    }
  }
}

// ---------------- flash attention ----------------
// grid = (T/64, H, N), block = 128; each wave owns 16 query rows, streams keys in
// chunks of 32 with online softmax. K and V fragments for the chunk are loaded
// up-front so V-load latency hides under the softmax. Writes concat layout
// [N][T][H*DH] in bf16.
__global__ void __launch_bounds__(128, 1)
flash_attn_kernel(const bf16_t* __restrict__ Q,
                  const bf16_t* __restrict__ K,
                  const bf16_t* __restrict__ Vt,
                  bf16_t* __restrict__ Cc) {
  __shared__ bf16_t pbuf[4][16][40];  // per-wave 16x32 P tile, padded stride
  const int wave = threadIdx.x >> 5;
  const int lane = threadIdx.x & 31;
  const int hf = lane >> 4;
  const int nl = lane & 15;
  const int h = blockIdx.y, n = blockIdx.z;
  const int q0 = blockIdx.x * 64 + wave * 16;

  const bf16_t* Qh = Q + ((size_t)(n * H_N + h)) * T_S * DH_;
  const bf16_t* Kh = K + ((size_t)(n * H_N + h)) * T_S * DH_;
  const bf16_t* Vh = Vt + ((size_t)(n * H_N + h)) * DH_ * T_S;

  // Q fragments for d = 0..31 and 32..63, loaded once
  v16bf aq0 = load_frag_a(Qh + (size_t)q0 * DH_ + 0, DH_);
  v16bf aq1 = load_frag_a(Qh + (size_t)q0 * DH_ + 32, DH_);

  v8f o0 = {}, o1 = {}, o2 = {}, o3 = {};
  float m[8], l[8];
#pragma unroll
  for (int r = 0; r < 8; ++r) { m[r] = -1e30f; l[r] = 0.f; }
  const float scale = 0.125f;  // 1/sqrt(DH)
  const int kend = q0 + 16;    // causal: keys 0 .. q0+15

  for (int kb = 0; kb < kend; kb += 32) {
    const bool t1 = (kb + 16) < kend;
    // ---- load all fragments for this chunk up front ----
    v16bf kf0 = load_frag_b(Kh + (size_t)kb * DH_ + 0, DH_);
    v16bf kf1 = load_frag_b(Kh + (size_t)kb * DH_ + 32, DH_);
    v16bf kf2 = kf0, kf3 = kf1;         // duplicates if tail tile absent (masked)
    if (t1) {
      kf2 = load_frag_b(Kh + (size_t)(kb + 16) * DH_ + 0, DH_);
      kf3 = load_frag_b(Kh + (size_t)(kb + 16) * DH_ + 32, DH_);
    }
    v16bf vb0 = load_frag_b(Vh + (size_t)0  * T_S + kb, T_S);
    v16bf vb1 = load_frag_b(Vh + (size_t)16 * T_S + kb, T_S);
    v16bf vb2 = load_frag_b(Vh + (size_t)32 * T_S + kb, T_S);
    v16bf vb3 = load_frag_b(Vh + (size_t)48 * T_S + kb, T_S);
    __builtin_prefetch(Kh + (size_t)(kb + 32 + lane) * DH_, 0, 3);

    // ---- scores ----
    v8f s0 = {}, s1 = {};
    s0 = wmma_bf16(aq0, kf0, s0);
    s0 = wmma_bf16(aq1, kf1, s0);
    s1 = wmma_bf16(aq0, kf2, s1);   // garbage when !t1; fully masked below
    s1 = wmma_bf16(aq1, kf3, s1);

    // ---- online softmax per row (VGPR r -> row M = r + 8*half) ----
#pragma unroll
    for (int r = 0; r < 8; ++r) {
      const int qg = q0 + r + hf * 8;
      float a = s0[r] * scale;
      if (kb + nl > qg) a = -1e30f;
      float b = s1[r] * scale;
      if (!t1 || (kb + 16 + nl > qg)) b = -1e30f;
      float mx = fmaxf(a, b);
      mx = fmaxf(mx, __shfl_xor(mx, 1));
      mx = fmaxf(mx, __shfl_xor(mx, 2));
      mx = fmaxf(mx, __shfl_xor(mx, 4));
      mx = fmaxf(mx, __shfl_xor(mx, 8));
      const float mn = fmaxf(m[r], mx);
      const float alpha = __expf(m[r] - mn);
      m[r] = mn;
      const float p0 = __expf(a - mn);
      const float p1 = __expf(b - mn);
      float rs = p0 + p1;
      rs += __shfl_xor(rs, 1);
      rs += __shfl_xor(rs, 2);
      rs += __shfl_xor(rs, 4);
      rs += __shfl_xor(rs, 8);
      l[r] = l[r] * alpha + rs;
      o0[r] *= alpha; o1[r] *= alpha; o2[r] *= alpha; o3[r] *= alpha;
      // C-layout -> A-layout staging through LDS
      pbuf[wave][r + hf * 8][nl]      = (bf16_t)p0;
      pbuf[wave][r + hf * 8][16 + nl] = (bf16_t)p1;
    }

    // ---- O += P(16x32) @ V(32x64) ----
    v16bf ap = load_frag_a(&pbuf[wave][0][0], 40);
    o0 = wmma_bf16(ap, vb0, o0);
    o1 = wmma_bf16(ap, vb1, o1);
    o2 = wmma_bf16(ap, vb2, o2);
    o3 = wmma_bf16(ap, vb3, o3);
  }

  // normalize and write concat[n][t][h*DH + d] (bf16)
  bf16_t* Cb = Cc + (size_t)n * T_S * E_D + (size_t)h * DH_;
#pragma unroll
  for (int r = 0; r < 8; ++r) {
    const float inv = 1.0f / l[r];
    const size_t row = (size_t)(q0 + r + hf * 8);
    Cb[row * E_D +  0 + nl] = (bf16_t)(o0[r] * inv);
    Cb[row * E_D + 16 + nl] = (bf16_t)(o1[r] * inv);
    Cb[row * E_D + 32 + nl] = (bf16_t)(o2[r] * inv);
    Cb[row * E_D + 48 + nl] = (bf16_t)(o3[r] * inv);
  }
}

// ---------------- output projection: Y = concat @ Wo^T + bo (fp32 out) ----------------
// grid = ((N*T)/128, E/64), block = 128; 32x64 per wave, ping-pong double-buffered.
__global__ void __launch_bounds__(128, 1)
out_proj_kernel(const bf16_t* __restrict__ Cc,
                const bf16_t* __restrict__ WoB,
                const float* __restrict__ bo,
                float* __restrict__ Y) {
  const int wave = threadIdx.x >> 5;
  const int lane = threadIdx.x & 31;
  const int hf = lane >> 4;
  const int nl = lane & 15;
  const int m0 = blockIdx.x * 128 + wave * 32;
  const int col0 = blockIdx.y * 64;

  const bf16_t* A0 = Cc + (size_t)m0 * E_D;
  const bf16_t* A1 = Cc + (size_t)(m0 + 16) * E_D;
  const bf16_t* W0 = WoB + (size_t)col0 * E_D;

  v8f acc0[4] = {{}, {}, {}, {}};
  v8f acc1[4] = {{}, {}, {}, {}};

  v16bf a0A = load_frag_a(A0, E_D);
  v16bf a1A = load_frag_a(A1, E_D);
  v16bf bA[4];
#pragma unroll
  for (int nt = 0; nt < 4; ++nt) bA[nt] = load_frag_b(W0 + (size_t)(nt * 16) * E_D, E_D);

  for (int e = 0; e < E_D; e += 64) {
    const int e1 = e + 32;
    v16bf a0B = load_frag_a(A0 + e1, E_D);
    v16bf a1B = load_frag_a(A1 + e1, E_D);
    v16bf bB[4];
#pragma unroll
    for (int nt = 0; nt < 4; ++nt)
      bB[nt] = load_frag_b(W0 + (size_t)(nt * 16) * E_D + e1, E_D);
#pragma unroll
    for (int nt = 0; nt < 4; ++nt) {
      acc0[nt] = wmma_bf16(a0A, bA[nt], acc0[nt]);
      acc1[nt] = wmma_bf16(a1A, bA[nt], acc1[nt]);
    }
    const int e2 = e + 64;
    if (e2 < E_D) {
      a0A = load_frag_a(A0 + e2, E_D);
      a1A = load_frag_a(A1 + e2, E_D);
#pragma unroll
      for (int nt = 0; nt < 4; ++nt)
        bA[nt] = load_frag_b(W0 + (size_t)(nt * 16) * E_D + e2, E_D);
    }
#pragma unroll
    for (int nt = 0; nt < 4; ++nt) {
      acc0[nt] = wmma_bf16(a0B, bB[nt], acc0[nt]);
      acc1[nt] = wmma_bf16(a1B, bB[nt], acc1[nt]);
    }
  }

#pragma unroll
  for (int nt = 0; nt < 4; ++nt) {
    const float bv = bo[col0 + nt * 16 + nl];
#pragma unroll
    for (int r = 0; r < 8; ++r) {
      const size_t r0 = (size_t)(m0 + r + hf * 8);
      Y[r0 * E_D + col0 + nt * 16 + nl]        = acc0[nt][r] + bv;
      Y[(r0 + 16) * E_D + col0 + nt * 16 + nl] = acc1[nt][r] + bv;
    }
  }
}

extern "C" void kernel_launch(void* const* d_in, const int* in_sizes, int n_in,
                              void* d_out, int out_size, void* d_ws, size_t ws_size,
                              hipStream_t stream) {
  (void)in_sizes; (void)n_in; (void)out_size; (void)ws_size;
  const float* query = (const float*)d_in[0];
  const float* keyin = (const float*)d_in[1];
  const float* valin = (const float*)d_in[2];
  // d_in[3] = causal mask [N,1,T,T]; causality is recomputed from indices (tril).
  const float* Wq = (const float*)d_in[4];
  const float* Wk = (const float*)d_in[5];
  const float* Wv = (const float*)d_in[6];
  const float* Wo = (const float*)d_in[7];
  const float* bo = (const float*)d_in[8];
  float* Y = (float*)d_out;

  char* ws = (char*)d_ws;
  size_t off = 0;
  auto alloc_bf = [&](size_t elems) {
    bf16_t* p = (bf16_t*)(ws + off);
    off += (elems * sizeof(bf16_t) + 255) & ~(size_t)255;
    return p;
  };
  const size_t nXE = (size_t)N_B * T_S * E_D;          // 4.19M elems
  const size_t nW  = (size_t)H_N * E_D * DH_;          // 1.05M elems
  const size_t nQ  = (size_t)N_B * H_N * T_S * DH_;    // 4.19M elems
  bf16_t* Xq  = alloc_bf(nXE);
  bf16_t* Xk  = alloc_bf(nXE);
  bf16_t* Xv  = alloc_bf(nXE);
  bf16_t* WqT = alloc_bf(nW);
  bf16_t* WkT = alloc_bf(nW);
  bf16_t* WvT = alloc_bf(nW);
  bf16_t* WoB = alloc_bf((size_t)E_D * E_D);
  bf16_t* Qb  = alloc_bf(nQ);
  bf16_t* Kb  = alloc_bf(nQ);
  bf16_t* Vt  = alloc_bf(nQ);
  bf16_t* Cc  = alloc_bf(nXE);   // ~64 MB total workspace

  const int nxe = (int)nXE, nw = (int)nW, nwo = E_D * E_D;
  cvt_f32_bf16<<<(nxe + 255) / 256, 256, 0, stream>>>(query, Xq, nxe);
  cvt_f32_bf16<<<(nxe + 255) / 256, 256, 0, stream>>>(keyin, Xk, nxe);
  cvt_f32_bf16<<<(nxe + 255) / 256, 256, 0, stream>>>(valin, Xv, nxe);
  cvt_w_transpose<<<(nw + 255) / 256, 256, 0, stream>>>(Wq, WqT);
  cvt_w_transpose<<<(nw + 255) / 256, 256, 0, stream>>>(Wk, WkT);
  cvt_w_transpose<<<(nw + 255) / 256, 256, 0, stream>>>(Wv, WvT);
  cvt_f32_bf16<<<(nwo + 255) / 256, 256, 0, stream>>>(Wo, WoB, nwo);

  dim3 pg(T_S / 128, H_N, N_B);
  proj_gemm_kernel<<<pg, 128, 0, stream>>>(Xq, WqT, Qb, 0);
  proj_gemm_kernel<<<pg, 128, 0, stream>>>(Xk, WkT, Kb, 0);
  proj_gemm_kernel<<<pg, 128, 0, stream>>>(Xv, WvT, Vt, 1);

  dim3 fg(T_S / 64, H_N, N_B);
  flash_attn_kernel<<<fg, 128, 0, stream>>>(Qb, Kb, Vt, Cc);

  dim3 og((N_B * T_S) / 128, E_D / 64);
  out_proj_kernel<<<og, 128, 0, stream>>>(Cc, WoB, bo, Y);
}
